// Cowan_network_65137474011853
// MI455X (gfx1250) — matrix operand order repaired and checked by reading, not compiled, not run on hardware
//
#include <hip/hip_runtime.h>
#include <hip/hip_bf16.h>
#include <math.h>

typedef __bf16 bf16_t;
typedef __bf16 v8bf  __attribute__((ext_vector_type(8)));
typedef __bf16 v16bf __attribute__((ext_vector_type(16)));
typedef float  v8f   __attribute__((ext_vector_type(8)));

#define SIZE   2048
#define NATTR  16
#define NTRAIN (SIZE - NATTR)   // 2032
#define BATCH  1024
#define TMAX   100

#define WEEc 7.2f
#define WEIc 2.0f
#define HEc  (-1.2f)
#define HIc  0.1f
#define AEc  1.5f
#define AIc  0.4f
#define FE1c 0.25f
#define FE2c 0.65f
#define FI1c 0.5f
#define FI2c 0.5f
#define B1c  3.7f
#define DTc  0.1f
#define DXc  0.022097086912079612f   // 1/sqrt(2048)

// ---------------------------------------------------------------------------
// CDNA5 async copy: global -> LDS, 16B per lane, tracked by ASYNCcnt.
// LDS address = low 32 bits of the generic pointer (flat aperture layout).
// ---------------------------------------------------------------------------
__device__ __forceinline__ void async_cp16(const bf16_t* lds_dst,
                                           const bf16_t* gsrc) {
  unsigned laddr = (unsigned)(unsigned long long)lds_dst;
  asm volatile("global_load_async_to_lds_b128 %0, %1, off"
               :: "v"(laddr), "v"(gsrc)
               : "memory");
}

__device__ __forceinline__ void wait_async0() {
#if __has_builtin(__builtin_amdgcn_s_wait_asynccnt)
  __builtin_amdgcn_s_wait_asynccnt(0);
#else
  asm volatile("s_wait_asynccnt 0x0" ::: "memory");
#endif
}

// ---------------------------------------------------------------------------
// Prep: recover Q from base (Q = base_train/1.15 | base_fix) and build
// W = Q * diag(clip(eig))  (fixed columns have eig = 0 -> W column = 0).
// Both stored bf16 row-major [r][t], t = eigvector index (contraction dim).
// ---------------------------------------------------------------------------
__global__ void __launch_bounds__(256)
prep_kernel(const float* __restrict__ base_train,
            const float* __restrict__ base_fix,
            const float* __restrict__ autov_tr,
            bf16_t* __restrict__ Qbf,
            bf16_t* __restrict__ Wbf) {
  int i = blockIdx.x * blockDim.x + threadIdx.x;   // over SIZE*SIZE
  if (i >= SIZE * SIZE) return;
  int r = i >> 11;       // / 2048
  int t = i & 2047;
  float q, w;
  if (t < NTRAIN) {
    q = base_train[(size_t)r * NTRAIN + t] / 1.15f;
    float e = autov_tr[t];
    e = fminf(fmaxf(e, -1000000.0f), 200.0f);
    w = q * e;
  } else {
    q = base_fix[(size_t)r * NATTR + (t - NTRAIN)];
    w = 0.0f;   // autov_fix == 0
  }
  Qbf[i] = (bf16_t)q;
  Wbf[i] = (bf16_t)w;
}

// ---------------------------------------------------------------------------
// Init state: X0 = Y0 = x, bf16 shadow of x.
// ---------------------------------------------------------------------------
__global__ void __launch_bounds__(256)
init_kernel(const float* __restrict__ x,
            float* __restrict__ X0,
            float* __restrict__ Y0,
            bf16_t* __restrict__ Xbf) {
  int i = blockIdx.x * blockDim.x + threadIdx.x;
  if (i >= BATCH * SIZE) return;
  float v = x[i];
  X0[i] = v;
  Y0[i] = v;
  Xbf[i] = (bf16_t)v;
}

// ---------------------------------------------------------------------------
// NT-GEMM: C[m,n] = alpha * sum_k A[m,k] * B[n,k]
// A: M x K row-major bf16, B: N x K row-major bf16 (contraction contiguous).
// Output: f32 row-major (outF) or bf16 (outBf if non-null).
// Block tile 128x128, BLK_K = 32, 256 threads = 8 waves (wave32).
// Double-buffered LDS filled by global_load_async_to_lds_b128; one barrier
// + one s_wait_asynccnt per K-step. Each wave: 2x4 grid of
// v_wmma_f32_16x16x32_bf16.
// ---------------------------------------------------------------------------
__global__ void __launch_bounds__(256)
gemm_nt_kernel(const bf16_t* __restrict__ A,
               const bf16_t* __restrict__ B,
               float* __restrict__ outF,
               bf16_t* __restrict__ outBf,
               int M, int N, int K, float alpha) {
  __shared__ __align__(16) bf16_t lA[2][128 * 32];
  __shared__ __align__(16) bf16_t lB[2][128 * 32];

  const int tid   = threadIdx.x;
  const int wave  = tid >> 5;
  const int lane  = tid & 31;
  const int waveM = wave >> 1;        // 0..3 -> rows waveM*32
  const int waveN = wave & 1;         // 0..1 -> cols waveN*64
  const int lrow  = lane & 15;
  const int lhalf = lane >> 4;

  const int bm = blockIdx.y * 128;
  const int bn = blockIdx.x * 128;

  // global tile fetch mapping: thread -> (row, 16-col segment), 2 x b128 each
  const int g_row = tid >> 1;          // 0..127
  const int g_seg = (tid & 1) * 16;    // 0 or 16

  const bf16_t* gA0 = A + (size_t)(bm + g_row) * K + g_seg;
  const bf16_t* gB0 = B + (size_t)(bn + g_row) * K + g_seg;

  // prologue: issue async fetch of tile 0 into buffer 0
  async_cp16(&lA[0][g_row * 32 + g_seg],     gA0);
  async_cp16(&lA[0][g_row * 32 + g_seg + 8], gA0 + 8);
  async_cp16(&lB[0][g_row * 32 + g_seg],     gB0);
  async_cp16(&lB[0][g_row * 32 + g_seg + 8], gB0 + 8);

  v8f acc[2][4] = {};
  const int nk = K >> 5;

  for (int kt = 0; kt < nk; ++kt) {
    const int buf = kt & 1;
    // publish tile kt: my async copies done, then workgroup barrier
    wait_async0();
    __syncthreads();

    // prefetch tile kt+1 into the other buffer (its last readers retired
    // at the barrier above)
    if (kt + 1 < nk) {
      const int nb = buf ^ 1;
      const bf16_t* gA = gA0 + (size_t)(kt + 1) * 32;
      const bf16_t* gB = gB0 + (size_t)(kt + 1) * 32;
      async_cp16(&lA[nb][g_row * 32 + g_seg],     gA);
      async_cp16(&lA[nb][g_row * 32 + g_seg + 8], gA + 8);
      async_cp16(&lB[nb][g_row * 32 + g_seg],     gB);
      async_cp16(&lB[nb][g_row * 32 + g_seg + 8], gB + 8);
    }

    // A-matrix fragments (16x32, MxK): lanes 0-15 K chunks {0..7,16..23},
    // lanes 16-31 K chunks {8..15,24..31}
    v16bf af[2];
#pragma unroll
    for (int m = 0; m < 2; ++m) {
      int row = waveM * 32 + m * 16 + lrow;
      v8bf lo = *(const v8bf*)(&lA[buf][row * 32 + lhalf * 8]);
      v8bf hi = *(const v8bf*)(&lA[buf][row * 32 + 16 + lhalf * 8]);
#pragma unroll
      for (int e = 0; e < 8; ++e) { af[m][e] = lo[e]; af[m][e + 8] = hi[e]; }
    }
    // B-matrix fragments (32x16, KxN): lanes 0-15 K chunks {0..7,16..23},
    // lanes 16-31 K chunks {8..15,24..31}
    v16bf bfr[4];
#pragma unroll
    for (int n = 0; n < 4; ++n) {
      int row = waveN * 64 + n * 16 + lrow;
      v8bf lo = *(const v8bf*)(&lB[buf][row * 32 + lhalf * 8]);
      v8bf hi = *(const v8bf*)(&lB[buf][row * 32 + 16 + lhalf * 8]);
#pragma unroll
      for (int e = 0; e < 8; ++e) { bfr[n][e] = lo[e]; bfr[n][e + 8] = hi[e]; }
    }

#pragma unroll
    for (int m = 0; m < 2; ++m)
#pragma unroll
      for (int n = 0; n < 4; ++n)
        acc[m][n] = __builtin_amdgcn_wmma_f32_16x16x32_bf16(
            false, af[m], false, bfr[n], (short)0, acc[m][n], false, false);
  }

  // D layout: VGPR r -> M = lhalf*8 + r, N = lrow
#pragma unroll
  for (int m = 0; m < 2; ++m) {
    int grow0 = bm + waveM * 32 + m * 16 + lhalf * 8;
#pragma unroll
    for (int n = 0; n < 4; ++n) {
      int gcol = bn + waveN * 64 + n * 16 + lrow;
#pragma unroll
      for (int r = 0; r < 8; ++r) {
        float v = acc[m][n][r] * alpha;
        size_t idx = (size_t)(grow0 + r) * N + gcol;
        if (outBf) outBf[idx] = (bf16_t)v;
        else       outF[idx]  = v;
      }
    }
  }
}

// ---------------------------------------------------------------------------
// One Wilson-Cowan step (elementwise). S already contains DX*(xs @ matrix_A).
// ---------------------------------------------------------------------------
__global__ void __launch_bounds__(256)
step_kernel(const float* __restrict__ S,
            const float* __restrict__ X,
            const float* __restrict__ Y,
            const float* __restrict__ gamma_p,
            float* __restrict__ Xo,
            float* __restrict__ Yo,
            bf16_t* __restrict__ Xbf) {
  int i = blockIdx.x * blockDim.x + threadIdx.x;
  if (i >= BATCH * SIZE) return;
  float xs = X[i], ys = Y[i];
  float I1 = WEEc * xs - WEIc * ys + HEc + S[i];
  float I2 = HIc - ys;                       // WIE = 0, WII = 1
  float fe = FE1c * tanhf(B1c * I1) + FE2c;
  float fi = FI1c * tanhf(I2) + FI2c;
  float xn = xs + DTc * (-AEc * xs + (1.0f - xs) * fe);
  float g  = gamma_p[0];
  float yn = ys + (DTc / g) * (-AIc * ys + (1.0f - ys) * fi);
  Xo[i] = xn;
  Yo[i] = yn;
  Xbf[i] = (bf16_t)xn;
}

// ---------------------------------------------------------------------------
extern "C" void kernel_launch(void* const* d_in, const int* in_sizes, int n_in,
                              void* d_out, int out_size, void* d_ws, size_t ws_size,
                              hipStream_t stream) {
  const float* x          = (const float*)d_in[0];
  const float* base_train = (const float*)d_in[1];
  const float* base_fix   = (const float*)d_in[2];
  const float* autov_tr   = (const float*)d_in[3];
  /* d_in[4] = autov_fix: all zeros, folded into prep */
  const float* gamma      = (const float*)d_in[5];

  char* p = (char*)d_ws;
  const size_t MB = (size_t)1 << 20;
  // persistent: A_dx bf16 [n][k] = DX * A[n,k]  (matrix_A contraction layout)
  bf16_t* Bmat = (bf16_t*)(p + 0 * MB);          // 8 MB
  // setup-only (aliased by state after setup GEMM):
  bf16_t* Qbf  = (bf16_t*)(p + 8 * MB);          // 8 MB  | X0
  float*  X0   = (float*) (p + 8 * MB);
  bf16_t* Wbf  = (bf16_t*)(p + 16 * MB);         // 8 MB  | Y0
  float*  Y0   = (float*) (p + 16 * MB);
  float*  X1   = (float*) (p + 24 * MB);         // 8 MB
  float*  Y1   = (float*) (p + 32 * MB);         // 8 MB
  float*  Sbuf = (float*) (p + 40 * MB);         // 8 MB
  bf16_t* xsbf = (bf16_t*)(p + 48 * MB);         // 4 MB   (total 52 MB)

  // 1) Q, W in bf16
  prep_kernel<<<(SIZE * SIZE) / 256, 256, 0, stream>>>(base_train, base_fix,
                                                       autov_tr, Qbf, Wbf);
  // 2) A_dx[n,k] = DX * sum_t W[n,t] * Q[k,t]   (2048^3 NT-GEMM, bf16 out)
  gemm_nt_kernel<<<dim3(SIZE / 128, SIZE / 128), 256, 0, stream>>>(
      Wbf, Qbf, nullptr, Bmat, SIZE, SIZE, SIZE, DXc);
  // 3) state init (overwrites Qbf/Wbf regions -- safe: stream-ordered)
  init_kernel<<<(BATCH * SIZE) / 256, 256, 0, stream>>>(x, X0, Y0, xsbf);

  // 4) 100 recurrence steps
  float *Xc = X0, *Yc = Y0, *Xn = X1, *Yn = Y1;
  for (int t = 0; t < TMAX; ++t) {
    // S[b,n] = sum_k xs_bf[b,k] * A_dx[n,k]
    gemm_nt_kernel<<<dim3(SIZE / 128, BATCH / 128), 256, 0, stream>>>(
        xsbf, Bmat, Sbuf, nullptr, BATCH, SIZE, SIZE, 1.0f);
    float* xo = (t == TMAX - 1) ? (float*)d_out : Xn;
    step_kernel<<<(BATCH * SIZE) / 256, 256, 0, stream>>>(
        Sbuf, Xc, Yc, gamma, xo, Yn, xsbf);
    float* tmp;
    tmp = Xc; Xc = Xn; Xn = tmp;
    tmp = Yc; Yc = Yn; Yn = tmp;
  }
}
